// LinearCRF_21062519620337
// MI455X (gfx1250) — matrix operand order repaired and checked by reading, not compiled, not run on hardware
//
#include <hip/hip_runtime.h>

typedef __attribute__((ext_vector_type(16))) _Float16 v16h;
typedef __attribute__((ext_vector_type(8)))  float    v8f;
typedef unsigned int u32x4 __attribute__((ext_vector_type(4)));
typedef int          i32x8 __attribute__((ext_vector_type(8)));
typedef int          i32x4 __attribute__((ext_vector_type(4)));

#define Bsz   256
#define Tlen  1024
#define Lnum  50
#define LP    64          // padded label dim for WMMA (K and N)
#define SIDX  47
#define EIDX  48
#define NEGI  (-1e30f)
#define GSTEP 8           // timesteps per TDM group
#define EMSTR (16 * Lnum) // floats per step-tile in LDS ring (16 rows x 50)

#if defined(__has_builtin)
#  if __has_builtin(__builtin_amdgcn_tensor_load_to_lds)
#    define HAVE_TDM 1
#  else
#    define HAVE_TDM 0
#  endif
#else
#  define HAVE_TDM 0
#endif

// Raw v_exp_f32 / v_log_f32 (operands kept in normal range -> no denorm fixups)
__device__ __forceinline__ float fexp(float x) {
    return __builtin_amdgcn_exp2f(x * 1.4426950408889634f);
}
__device__ __forceinline__ float flog(float x) {
    return __builtin_amdgcn_logf(x) * 0.6931471805599453f;
}
__device__ __forceinline__ unsigned lds_off(const void* p) {
    return (unsigned)(unsigned long long)p;   // LDS aperture: addr[31:0] = offset
}
__device__ __forceinline__ int imin(int a, int b) { return a < b ? a : b; }

__global__ void crf_zero_out(float* out) {
    out[0] = 0.0f;
    out[1] = 0.0f;
}

#if HAVE_TDM
// Issue one TDM op loading `cnt` consecutive step-tiles (16 rows x 50 f32,
// row stride T*L) into LDS at lds_byte, compacted every EMSTR floats via
// descriptor iteration.  D# packing per CDNA5 ISA.
// Toolchain uses the 6-arg builtin: (g0, g1, g2, g3, g4, cpol).
__device__ __forceinline__ void tdm_load_emit(const float* gsrc,
                                              unsigned lds_byte, int cnt) {
    unsigned long long ga = (unsigned long long)gsrc;
    u32x4 g0;
    g0[0] = 1u;                                     // count=1, user descriptor
    g0[1] = lds_byte;                               // lds_addr
    g0[2] = (unsigned)(ga & 0xffffffffu);           // global_addr[31:0]
    g0[3] = (unsigned)((ga >> 32) & 0x01ffffffu)    // global_addr[56:32]
          | (2u << 30);                             // type=2 ("image")
    i32x8 g1;
    g1[0] = (2 << 16) | (1 << 19);                  // data_size=4B, iterate_en
    g1[1] = (Lnum << 16);                           // tensor_dim0[15:0]=50
    g1[2] = (16 << 16);                             // tensor_dim1[15:0]=16
    g1[3] = (Lnum << 16);                           // tile_dim0=50
    g1[4] = 16;                                     // tile_dim1=16, tile_dim2=0
    g1[5] = Tlen * Lnum;                            // tensor_dim0_stride=51200
    g1[6] = 0;
    g1[7] = 0;
    i32x4 g2;
    g2[0] = 0;                                      // tensor_dim2
    g2[1] = EMSTR;                                  // lds_addr_increment (elems)
    g2[2] = Lnum;                                   // global_addr_increment
    g2[3] = (cnt - 1) << 16;                        // iterate_count (0 = 1x)
    i32x4 g3 = {0, 0, 0, 0};
    i32x8 g4 = {0, 0, 0, 0, 0, 0, 0, 0};
    __builtin_amdgcn_tensor_load_to_lds(g0, g1, g2, g3, g4, 0);
}
#endif

// ---------------------------------------------------------------------------
// Forward (partition function). One wave per 16-row batch tile.
// alpha_new = amax + colmax + log( U x P ) + emit, U = exp(alpha-amax) (A,f16),
// P = exp(trans-colmax) (B, f16, constant over t, VGPR-resident).
// 8x v_wmma_f32_16x16x32_f16 per step; emissions streamed by the TDM.
// ---------------------------------------------------------------------------
__global__ __launch_bounds__(32)
void crf_forward_kernel(const float* __restrict__ lstm,
                        const float* __restrict__ trans,
                        const int*   __restrict__ lens,
                        float*       __restrict__ out)
{
    __shared__ float s_alpha[16 * LP];        // [b][j], -1e30 padded
    __shared__ float s_amax[16];
    __shared__ float s_cmax[LP];
#if HAVE_TDM
    __shared__ float s_emit[2][GSTEP * EMSTR];   // double-buffered TDM ring
#endif

    const int lane = threadIdx.x;             // wave32
    const int half = (lane >= 16) ? 1 : 0;
    const int n    = lane & 15;
    const int bg0  = blockIdx.x * 16;

    // ---- column max of transition (padding cols -> 0) ----
    for (int jj = lane; jj < LP; jj += 32) {
        float m = 0.0f;
        if (jj < Lnum) {
            m = -3.0e38f;
            for (int i = 0; i < Lnum; ++i) m = fmaxf(m, trans[i * Lnum + jj]);
        }
        s_cmax[jj] = m;
    }
    __syncthreads();

    float cm[4];
#pragma unroll
    for (int jt = 0; jt < 4; ++jt) cm[jt] = s_cmax[jt * 16 + n];

    // ---- constant B fragments: P[i,j] = exp(trans[i,j]-colmax[j]) ----
    // B (32x16 f16): lane&15 = N; element e -> K = e (+16 for lanes>=16)
    v16h bfrag[4][2];
#pragma unroll
    for (int jt = 0; jt < 4; ++jt) {
        const int j = jt * 16 + n;
#pragma unroll
        for (int kc = 0; kc < 2; ++kc) {
            v16h f;
#pragma unroll
            for (int e = 0; e < 16; ++e) {
                const int i = kc * 32 + half * 16 + e;
                float v = 0.0f;
                if (i < Lnum && j < Lnum) v = fexp(trans[i * Lnum + j] - cm[jt]);
                f[e] = (_Float16)v;
            }
            bfrag[jt][kc] = f;
        }
    }

    // ---- t = 0 ----
    for (int idx = lane; idx < 16 * LP; idx += 32) {
        const int b = idx >> 6, j = idx & 63;
        float a = NEGI;
        if (j < Lnum)
            a = trans[SIDX * Lnum + j] + lstm[((size_t)(bg0 + b) * Tlen) * Lnum + j];
        s_alpha[idx] = a;
    }
    __syncthreads();
    if (lane < 16) {
        float m = NEGI;
        for (int j = 0; j < Lnum; ++j) m = fmaxf(m, s_alpha[lane * LP + j]);
        s_amax[lane] = m;
    }
    __syncthreads();

    float unl   = 0.0f;
    int   mylen = (lane < 16) ? lens[bg0 + lane] : -1;

    if (lane < 16 && mylen - 1 == 0) {
        float m = NEGI;
        for (int j = 0; j < Lnum; ++j)
            m = fmaxf(m, s_alpha[lane * LP + j] + trans[j * Lnum + EIDX]);
        float s = 0.0f;
        for (int j = 0; j < Lnum; ++j)
            s += fexp(s_alpha[lane * LP + j] + trans[j * Lnum + EIDX] - m);
        unl += m + flog(s);
    }

    // running per-row max carried in registers (rows b = v + half*8)
    float pam[8];
#pragma unroll
    for (int v = 0; v < 8; ++v) pam[v] = s_amax[v + half * 8];
    __syncthreads();

    const int ngroups = (Tlen - 1 + GSTEP - 1) / GSTEP;   // steps t=1..T-1
#if HAVE_TDM
    tdm_load_emit(lstm + ((size_t)bg0 * Tlen + 1) * Lnum, lds_off(&s_emit[0][0]),
                  imin(GSTEP, Tlen - 1));
#endif

    for (int g = 0; g < ngroups; ++g) {
        const int tstart = 1 + g * GSTEP;
        const int cnt    = imin(GSTEP, Tlen - tstart);
#if HAVE_TDM
        if (g + 1 < ngroups) {
            const int ts2 = 1 + (g + 1) * GSTEP;
            tdm_load_emit(lstm + ((size_t)bg0 * Tlen + ts2) * Lnum,
                          lds_off(&s_emit[(g + 1) & 1][0]),
                          imin(GSTEP, Tlen - ts2));
            __builtin_amdgcn_s_wait_tensorcnt(1);   // current ring resident
        } else {
            __builtin_amdgcn_s_wait_tensorcnt(0);
        }
        __syncthreads();
        const float* eb = &s_emit[g & 1][0];
#endif
        for (int k = 0; k < cnt; ++k) {
            const int t = tstart + k;
            // A fragments: lane&15 = M; e -> K = e + (half?16:0)
            const float am = s_amax[n];
            v16h a0, a1;
#pragma unroll
            for (int e = 0; e < 16; ++e) {
                const int K  = e + (half ? 16 : 0);
                const int i0 = K, i1 = 32 + K;
                float v0 = (i0 < Lnum) ? fexp(s_alpha[n * LP + i0] - am) : 0.0f;
                float v1 = (i1 < Lnum) ? fexp(s_alpha[n * LP + i1] - am) : 0.0f;
                a0[e] = (_Float16)v0;
                a1[e] = (_Float16)v1;
            }
            __syncthreads();          // alpha/amax reads done before overwrite

            float mloc[8];
#pragma unroll
            for (int v = 0; v < 8; ++v) mloc[v] = NEGI;

#pragma unroll
            for (int jt = 0; jt < 4; ++jt) {
                v8f c = {};
                c = __builtin_amdgcn_wmma_f32_16x16x32_f16(
                        false, a0, false, bfrag[jt][0], (short)0, c, false, false);
                c = __builtin_amdgcn_wmma_f32_16x16x32_f16(
                        false, a1, false, bfrag[jt][1], (short)0, c, false, false);
                const int j = jt * 16 + n;
#pragma unroll
                for (int v = 0; v < 8; ++v) {
                    const int b  = v + half * 8;
                    float     na = NEGI;
                    if (j < Lnum) {
#if HAVE_TDM
                        const float ev = eb[k * EMSTR + b * Lnum + j];
#else
                        const float ev =
                            lstm[((size_t)(bg0 + b) * Tlen + t) * Lnum + j];
#endif
                        na = pam[v] + cm[jt] + flog(fmaxf(c[v], 1e-37f)) + ev;
                        mloc[v] = fmaxf(mloc[v], na);
                    }
                    s_alpha[b * LP + j] = na;
                }
            }

            // per-row max across the 16 lanes of each half
#pragma unroll
            for (int off = 8; off >= 1; off >>= 1) {
#pragma unroll
                for (int v = 0; v < 8; ++v)
                    mloc[v] = fmaxf(mloc[v], __shfl_xor(mloc[v], off, 32));
            }
#pragma unroll
            for (int v = 0; v < 8; ++v) pam[v] = mloc[v];
            if (n == 0) {
#pragma unroll
                for (int v = 0; v < 8; ++v) s_amax[v + half * 8] = mloc[v];
            }
            __syncthreads();          // alpha/amax writes visible

            if (lane < 16 && mylen - 1 == t) {
                float m = NEGI;
                for (int j = 0; j < Lnum; ++j)
                    m = fmaxf(m, s_alpha[lane * LP + j] + trans[j * Lnum + EIDX]);
                float s = 0.0f;
                for (int j = 0; j < Lnum; ++j)
                    s += fexp(s_alpha[lane * LP + j] + trans[j * Lnum + EIDX] - m);
                unl += m + flog(s);
            }
        }
    }

    if (lane < 16) atomicAdd(out + 0, unl);
}

// ---------------------------------------------------------------------------
// Labeled (gold path) score: one block per batch row, gather + reduce.
// ---------------------------------------------------------------------------
__global__ __launch_bounds__(256)
void crf_labeled_kernel(const float* __restrict__ lstm,
                        const float* __restrict__ trans,
                        const int*   __restrict__ lens,
                        const int*   __restrict__ tags,
                        float*       __restrict__ out)
{
    const int  b   = blockIdx.x;
    const int  tid = threadIdx.x;
    const int  len = lens[b];
    const int* tg  = tags + (size_t)b * Tlen;

    float s = 0.0f;
    for (int t = 1 + tid; t < Tlen; t += 256) {
        if (t < len) {
            const int prev = tg[t - 1], cur = tg[t];
            s += trans[prev * Lnum + cur] + lstm[((size_t)b * Tlen + t) * Lnum + cur];
        }
    }

    __shared__ float red[256];
    red[tid] = s;
    __syncthreads();
    for (int off = 128; off >= 1; off >>= 1) {
        if (tid < off) red[tid] += red[tid + off];
        __syncthreads();
    }
    if (tid == 0) {
        const float begin =
            trans[SIDX * Lnum + tg[0]] + lstm[((size_t)b * Tlen) * Lnum + tg[0]];
        const float endv = trans[tg[len - 1] * Lnum + EIDX];
        atomicAdd(out + 1, red[0] + begin + endv);
    }
}

extern "C" void kernel_launch(void* const* d_in, const int* in_sizes, int n_in,
                              void* d_out, int out_size, void* d_ws, size_t ws_size,
                              hipStream_t stream) {
    (void)in_sizes; (void)n_in; (void)out_size; (void)d_ws; (void)ws_size;
    const float* lstm  = (const float*)d_in[0];   // (B,T,L) f32
    const float* trans = (const float*)d_in[1];   // (L,L)   f32
    const int*   lens  = (const int*)d_in[2];     // (B,)    i32
    const int*   tags  = (const int*)d_in[3];     // (B,T)   i32
    float*       out   = (float*)d_out;

    crf_zero_out<<<dim3(1), dim3(1), 0, stream>>>(out);
    crf_forward_kernel<<<dim3(Bsz / 16), dim3(32), 0, stream>>>(lstm, trans, lens, out);
    crf_labeled_kernel<<<dim3(Bsz), dim3(256), 0, stream>>>(lstm, trans, lens, tags, out);
}
